// CTRGC_48765058679545
// MI455X (gfx1250) — matrix operand order repaired and checked by reading, not compiled, not run on hardware
//
#include <hip/hip_runtime.h>
#include <hip/hip_bf16.h>
#include <math.h>

#define N_B    128
#define C_IN   64
#define C_OUT  64
#define T_DIM  64
#define V_DIM  25
#define SP     (T_DIM * V_DIM)   // 1600 spatial positions per (n, channel)
#define RELN   16
#define CH_ALL 320               // 64 (x3) + 128 (x_a) + 128 (x_b)

typedef __attribute__((ext_vector_type(16))) __bf16 v16bf;
typedef __attribute__((ext_vector_type(8)))  float  v8f;
typedef __attribute__((address_space(3))) unsigned short lds_us;

union Frag16 { v16bf v; unsigned short u[16]; };

// fp32 -> bf16 (round to nearest even, bit-level)
static __device__ __forceinline__ unsigned short f2bf(float f) {
  unsigned int u = __builtin_bit_cast(unsigned int, f);
  u += 0x7fffu + ((u >> 16) & 1u);
  return (unsigned short)(u >> 16);
}

static __device__ __forceinline__ v8f wmma_bf16(const Frag16 &a, const Frag16 &b, v8f c) {
  // D = A(16x32 bf16) * B(32x16 bf16) + C(16x16 f32)
  return __builtin_amdgcn_wmma_f32_16x16x32_bf16(false, a.v, false, b.v, (short)0, c,
                                                 false, false);
}

// K-index for element e of a 16-bit A-matrix fragment, lane half hf.
// K = 16*hf + e (contiguous order, matches the B fragments)
static __device__ __forceinline__ int a_k(int e, int hf) {
  return 16 * hf + e;
}

// CDNA5 async global -> LDS copy (16B per lane), tracked by ASYNCcnt.
static __device__ __forceinline__ void async_g2l_b128(lds_us* lptr, const unsigned short* g) {
  asm volatile("global_load_async_to_lds_b128 %0, %1, off" : : "v"(lptr), "v"(g) : "memory");
}
static __device__ __forceinline__ void wait_async0() {
  asm volatile("s_wait_asynccnt 0x0" : : : "memory");
}

// ---------------------------------------------------------------------------
// K1: fused 1x1 convs (w3 | wa | wb) over x.  GEMM: [16 spatial] x [K=64] x
// [320 out channels].  One wave per spatial tile; x tile staged in LDS
// TRANSPOSED ([m][cin]) and pre-converted to bf16 so both A and B fragment
// builds are contiguous 16B LDS reads.  Weights staged once per block (bf16).
// Output y_all: (N, 320, 1600) bf16 — ch 0..63 = x3, 64..191 = x_a, 192..319 = x_b.
// ---------------------------------------------------------------------------
__global__ __launch_bounds__(128)
void k1_conv_abc(const float* __restrict__ x,
                 const float* __restrict__ w3, const float* __restrict__ b3,
                 const float* __restrict__ wa, const float* __restrict__ ba,
                 const float* __restrict__ wb, const float* __restrict__ bb,
                 unsigned short* __restrict__ y_all) {
  __shared__ unsigned short Wl[CH_ALL * C_IN];    // 40 KB bf16 weights [o][c]
  __shared__ float          Bl[CH_ALL];
  __shared__ unsigned short Xl[4][16 * C_IN];     // per-wave x tile, [m][c] bf16
  const int tid = threadIdx.x;
  for (int i = tid; i < CH_ALL * C_IN; i += 128) {
    const int o = i >> 6, c = i & 63;
    const float w = (o < 64) ? w3[o * 64 + c]
                  : (o < 192) ? wa[(o - 64) * 64 + c]
                              : wb[(o - 192) * 64 + c];
    Wl[i] = f2bf(w);
  }
  for (int i = tid; i < CH_ALL; i += 128)
    Bl[i] = (i < 64) ? b3[i] : (i < 192) ? ba[i - 64] : bb[i - 192];
  __syncthreads();

  const int n = blockIdx.x;
  const int wv = tid >> 5, lane = tid & 31;
  const int hf = lane >> 4, m = lane & 15;
  const float* xp = x + (size_t)n * C_IN * SP;
  unsigned short* yp = y_all + (size_t)n * CH_ALL * SP;
  unsigned short* xs = Xl[wv];

  for (int it = 0; it < 5; ++it) {
    const int s0 = (blockIdx.y * 20 + it * 4 + wv) * 16;
    for (int i = lane; i < C_IN * 16; i += 32) {
      const int c = i >> 4, sl = i & 15;
      xs[sl * 64 + c] = f2bf(xp[(size_t)c * SP + s0 + sl]);   // transpose to [m][c]
    }
    __syncthreads();
    Frag16 a0, a1;                                            // K = cin 0..31, 32..63
#pragma unroll
    for (int e = 0; e < 16; ++e) {
      const int K = a_k(e, hf);                               // contiguous 16-runs
      a0.u[e] = xs[m * 64 + K];
      a1.u[e] = xs[m * 64 + 32 + K];
    }
    for (int ot = 0; ot < 20; ++ot) {
      const int oc = ot * 16 + m;                             // B column = lane%16
      Frag16 bf0, bf1;
#pragma unroll
      for (int e = 0; e < 16; ++e) {
        const int K = 16 * hf + e;
        bf0.u[e] = Wl[oc * 64 + K];
        bf1.u[e] = Wl[oc * 64 + 32 + K];
      }
      v8f acc = {};
      acc = wmma_bf16(a0, bf0, acc);
      acc = wmma_bf16(a1, bf1, acc);
      const float bias = Bl[oc];
#pragma unroll
      for (int i = 0; i < 8; ++i)
        yp[(size_t)oc * SP + s0 + i + 8 * hf] = f2bf(acc[i] + bias);
    }
    __syncthreads();
  }
}

// ---------------------------------------------------------------------------
// K2: M[n,r,u,v] = tanh( (1/512) * sum_{j<8,t<64} xa[n,r*8+j,t,u]*xb[n,r*8+j,t,v] )
// One block (4 waves) per (n, r).  Operands staged in LDS TRANSPOSED to
// [u][j*64+t] / [v][j*64+t] so fragment reads are contiguous; each wave owns
// one 16x16 output tile of the (padded) 32x32 result.  K=512 in 16 steps.
// ---------------------------------------------------------------------------
__global__ __launch_bounds__(128)
void k2_rel_mat(const unsigned short* __restrict__ y_all, float* __restrict__ Mb) {
  __shared__ unsigned short sa[V_DIM * 512];   // [u][j*64+t]  25.6 KB
  __shared__ unsigned short sb[V_DIM * 512];   // [v][j*64+t]  25.6 KB
  const int n = blockIdx.x, r = blockIdx.y;
  const int tid = threadIdx.x;
  const unsigned short* ya = y_all + (size_t)n * CH_ALL * SP + (size_t)(64  + r * 8) * SP;
  const unsigned short* yb = y_all + (size_t)n * CH_ALL * SP + (size_t)(192 + r * 8) * SP;
  for (int i = tid; i < 8 * SP; i += 128) {
    const int j = i / SP, rem = i - j * SP;
    const int t = rem / V_DIM, v = rem - t * V_DIM;
    sa[v * 512 + j * 64 + t] = ya[i];          // coalesced read, transposed store
    sb[v * 512 + j * 64 + t] = yb[i];
  }
  __syncthreads();

  const int wv = tid >> 5, lane = tid & 31;
  const int hf = lane >> 4, m = lane & 15;
  const int ut = wv >> 1, vt = wv & 1;
  const int u = ut * 16 + m;                   // A row (joint index u)
  const int v = vt * 16 + m;                   // B column (joint index v)
  v8f acc = {};
  for (int kk = 0; kk < 16; ++kk) {
    const int j = kk >> 1;
    const int tb = (kk & 1) * 32;
    const int base = j * 64 + tb;
    Frag16 a, b;
#pragma unroll
    for (int e = 0; e < 16; ++e) {
      const int K = a_k(e, hf);
      a.u[e] = (u < V_DIM) ? sa[u * 512 + base + K] : (unsigned short)0;
      b.u[e] = (v < V_DIM) ? sb[v * 512 + base + 16 * hf + e] : (unsigned short)0;
    }
    acc = wmma_bf16(a, b, acc);
  }
  float* Mp = Mb + ((size_t)n * RELN + r) * (V_DIM * V_DIM);
  const float sc = 1.0f / (float)(8 * T_DIM);
  if (v < V_DIM) {
#pragma unroll
    for (int i = 0; i < 8; ++i) {
      const int uu = ut * 16 + i + 8 * hf;
      if (uu < V_DIM) Mp[uu * V_DIM + v] = tanhf(acc[i] * sc);
    }
  }
}

// ---------------------------------------------------------------------------
// K3: x_1[n,o,u,v] = (sum_r wA[o,r]*M[n,r,u,v] + bA[o]) * alpha + A[u,v]  (bf16 out)
// ---------------------------------------------------------------------------
__global__ __launch_bounds__(256)
void k3_mix_adj(const float* __restrict__ Mb, const float* __restrict__ wA,
                const float* __restrict__ bA, const float* __restrict__ Aadj,
                unsigned short* __restrict__ x1q) {
  const int idx = blockIdx.x * 256 + threadIdx.x;
  const int total = N_B * C_OUT * V_DIM * V_DIM;
  if (idx >= total) return;
  const int uv = idx % 625;
  const int o  = (idx / 625) % C_OUT;
  const int n  = idx / (625 * C_OUT);
  const float* Mn = Mb + (size_t)n * RELN * 625;
  float acc = bA[o];
#pragma unroll
  for (int r = 0; r < RELN; ++r) acc += wA[o * RELN + r] * Mn[r * 625 + uv];
  acc += Aadj[uv];                          // alpha == 1.0
  x1q[idx] = f2bf(acc);
}

// ---------------------------------------------------------------------------
// K4: x1pre[n,c,t,u] = sum_v x_1[n,c,u,v] * x3[n,c,t,v]   (A: 64x25, B: x_1^T)
// One wave per (n,c).  x3 tile (3200 B) staged to LDS via CDNA5 async
// global->LDS b128 (ASYNCcnt); x_1 tile staged normally.  4 t-tiles x 2
// u-tiles, single K-step (v padded 25->32).
// ---------------------------------------------------------------------------
__global__ __launch_bounds__(32)
void k4_graph_apply(const unsigned short* __restrict__ y_all,
                    const unsigned short* __restrict__ x1q,
                    unsigned short* __restrict__ x1pre) {
  __shared__ unsigned short s3[SP];        // x3[t][v]  1600 us = 3200 B (200 x 16B)
  __shared__ unsigned short sq[640];       // x_1[u][v] 625 us
  const int n = blockIdx.x, c = blockIdx.y;
  const int lane = threadIdx.x, hf = lane >> 4, m = lane & 15;
  const unsigned short* x3p = y_all + (size_t)n * CH_ALL * SP + (size_t)c * SP;
  const unsigned short* q   = x1q + ((size_t)n * C_OUT + c) * 625;
  unsigned short* op = x1pre + ((size_t)n * C_OUT + c) * SP;

  lds_us* l3 = (lds_us*)s3;
  for (int i = lane; i < 200; i += 32)                     // async DMA: 200 x 16B
    async_g2l_b128(l3 + i * 8, x3p + i * 8);
  for (int i = lane; i < 625; i += 32) sq[i] = q[i];
  wait_async0();
  __syncthreads();

  Frag16 b0, b1;                             // B[v][u] = x_1[u][v]
#pragma unroll
  for (int e = 0; e < 16; ++e) {
    const int v = 16 * hf + e;
    b0.u[e] = (v < V_DIM) ? sq[m * V_DIM + v] : (unsigned short)0;
    b1.u[e] = (v < V_DIM && 16 + m < V_DIM) ? sq[(16 + m) * V_DIM + v] : (unsigned short)0;
  }
  for (int tt = 0; tt < 4; ++tt) {
    Frag16 a;                                // A[t][v] = x3[t][v]
#pragma unroll
    for (int e = 0; e < 16; ++e) {
      const int v = a_k(e, hf);
      a.u[e] = (v < V_DIM) ? s3[(tt * 16 + m) * V_DIM + v] : (unsigned short)0;
    }
    v8f c0 = {}, c1 = {};
    c0 = wmma_bf16(a, b0, c0);
    c1 = wmma_bf16(a, b1, c1);
#pragma unroll
    for (int i = 0; i < 8; ++i) {
      const int t = tt * 16 + i + 8 * hf;
      op[t * V_DIM + m] = f2bf(c0[i]);
      if (16 + m < V_DIM) op[t * V_DIM + 16 + m] = f2bf(c1[i]);
    }
  }
}

// ---------------------------------------------------------------------------
// K5: x1 = w4 @ x1pre + b4  (bf16 in, f32 out -> d_out); same transposed-LDS
// staging trick as K1.
// ---------------------------------------------------------------------------
__global__ __launch_bounds__(128)
void k5_conv4(const unsigned short* __restrict__ x1pre,
              const float* __restrict__ w4, const float* __restrict__ b4,
              float* __restrict__ out) {
  __shared__ unsigned short Wl[C_OUT * C_OUT];
  __shared__ float          Bl[C_OUT];
  __shared__ unsigned short Xl[4][16 * C_OUT];   // [m][c] bf16
  const int tid = threadIdx.x;
  for (int i = tid; i < C_OUT * C_OUT; i += 128) Wl[i] = f2bf(w4[i]);
  for (int i = tid; i < C_OUT; i += 128) Bl[i] = b4[i];
  __syncthreads();
  const int n = blockIdx.x;
  const int wv = tid >> 5, lane = tid & 31, hf = lane >> 4, m = lane & 15;
  const unsigned short* xp = x1pre + (size_t)n * C_OUT * SP;
  float* yp = out + (size_t)n * C_OUT * SP;
  unsigned short* xs = Xl[wv];
  for (int it = 0; it < 5; ++it) {
    const int s0 = (blockIdx.y * 20 + it * 4 + wv) * 16;
    for (int i = lane; i < C_OUT * 16; i += 32) {
      const int c = i >> 4, sl = i & 15;
      xs[sl * 64 + c] = xp[(size_t)c * SP + s0 + sl];
    }
    __syncthreads();
    Frag16 a0, a1;
#pragma unroll
    for (int e = 0; e < 16; ++e) {
      const int K = a_k(e, hf);
      a0.u[e] = xs[m * 64 + K];
      a1.u[e] = xs[m * 64 + 32 + K];
    }
    for (int ot = 0; ot < 4; ++ot) {
      const int oc = ot * 16 + m;
      Frag16 bf0, bf1;
#pragma unroll
      for (int e = 0; e < 16; ++e) {
        const int K = 16 * hf + e;
        bf0.u[e] = Wl[oc * 64 + K];
        bf1.u[e] = Wl[oc * 64 + 32 + K];
      }
      v8f acc = {};
      acc = wmma_bf16(a0, bf0, acc);
      acc = wmma_bf16(a1, bf1, acc);
      const float bias = Bl[oc];
#pragma unroll
      for (int i = 0; i < 8; ++i)
        yp[(size_t)oc * SP + s0 + i + 8 * hf] = acc[i] + bias;
    }
    __syncthreads();
  }
}

// ---------------------------------------------------------------------------
// K6: SE squeeze-excite: per-n channel means -> relu(se_w1) -> sigmoid(se_w2)
// ---------------------------------------------------------------------------
__global__ __launch_bounds__(256)
void k6_se(const float* __restrict__ x1, const float* __restrict__ se_w1,
           const float* __restrict__ se_w2, float* __restrict__ sbuf) {
  __shared__ float part[256];
  __shared__ float ym[C_OUT];
  __shared__ float y8[8];
  const int n = blockIdx.x, tid = threadIdx.x;
  const int c = tid >> 2, q = tid & 3;
  const float* xp = x1 + (size_t)n * C_OUT * SP + (size_t)c * SP;
  float s = 0.f;
  for (int i = q * 400; i < q * 400 + 400; ++i) s += xp[i];
  part[tid] = s;
  __syncthreads();
  if (q == 0) ym[c] = (part[tid] + part[tid + 1] + part[tid + 2] + part[tid + 3]) * (1.0f / (float)SP);
  __syncthreads();
  if (tid < 8) {
    float a = 0.f;
    for (int cc = 0; cc < C_OUT; ++cc) a += se_w1[tid * C_OUT + cc] * ym[cc];
    y8[tid] = fmaxf(a, 0.f);
  }
  __syncthreads();
  if (tid < C_OUT) {
    float a = 0.f;
#pragma unroll
    for (int j = 0; j < 8; ++j) a += se_w2[tid * 8 + j] * y8[j];
    sbuf[n * C_OUT + tid] = 1.0f / (1.0f + expf(-a));
  }
}

// K7: out *= s[n,c]  (in place; out fully rewritten by K5 each call)
__global__ __launch_bounds__(256)
void k7_scale(float* __restrict__ out, const float* __restrict__ sbuf) {
  const int idx = blockIdx.x * 256 + threadIdx.x;
  const int total = N_B * C_OUT * SP;
  if (idx >= total) return;
  const int c = (idx / SP) % C_OUT;
  const int n = idx / (C_OUT * SP);
  out[idx] *= sbuf[n * C_OUT + c];
}

extern "C" void kernel_launch(void* const* d_in, const int* in_sizes, int n_in,
                              void* d_out, int out_size, void* d_ws, size_t ws_size,
                              hipStream_t stream) {
  const float* x    = (const float*)d_in[0];
  const float* Aadj = (const float*)d_in[1];
  const float* w3   = (const float*)d_in[2];
  const float* b3   = (const float*)d_in[3];
  const float* wa   = (const float*)d_in[4];
  const float* ba   = (const float*)d_in[5];
  const float* wb   = (const float*)d_in[6];
  const float* bb   = (const float*)d_in[7];
  const float* wA   = (const float*)d_in[8];
  const float* bA   = (const float*)d_in[9];
  const float* w4   = (const float*)d_in[10];
  const float* b4   = (const float*)d_in[11];
  const float* sw1  = (const float*)d_in[12];
  const float* sw2  = (const float*)d_in[13];
  float* out = (float*)d_out;

  char* ws = (char*)d_ws;
  size_t off = 0;
  unsigned short* y_all = (unsigned short*)(ws + off); off += (size_t)N_B * CH_ALL * SP * 2;
  float*          Mb    = (float*)(ws + off);          off += (size_t)N_B * RELN * 625 * 4;
  unsigned short* x1q   = (unsigned short*)(ws + off); off += (size_t)N_B * C_OUT * 625 * 2;
  unsigned short* x1pre = (unsigned short*)(ws + off); off += (size_t)N_B * C_OUT * SP * 2;
  float*          sbuf  = (float*)(ws + off);

  k1_conv_abc<<<dim3(N_B, 5), 128, 0, stream>>>(x, w3, b3, wa, ba, wb, bb, y_all);
  k2_rel_mat<<<dim3(N_B, RELN), 128, 0, stream>>>(y_all, Mb);
  {
    const int total = N_B * C_OUT * 625;
    k3_mix_adj<<<(total + 255) / 256, 256, 0, stream>>>(Mb, wA, bA, Aadj, x1q);
  }
  k4_graph_apply<<<dim3(N_B, C_OUT), 32, 0, stream>>>(y_all, x1q, x1pre);
  k5_conv4<<<dim3(N_B, 5), 128, 0, stream>>>(x1pre, w4, b4, out);
  k6_se<<<N_B, 256, 0, stream>>>(out, sw1, sw2, sbuf);
  {
    const int total = N_B * C_OUT * SP;
    k7_scale<<<(total + 255) / 256, 256, 0, stream>>>(out, sbuf);
  }
}
